// UpSample_22436909154643
// MI455X (gfx1250) — matrix operand order repaired and checked by reading, not compiled, not run on hardware
//
#include <hip/hip_runtime.h>

// ---------------------------------------------------------------------------
// MI455X (gfx1250) fused attention for the UpSample reference.
//   B=4, C=128, N=8192, M=2048.
// Flash-attention style: energy [B,N,M] (268 MB) is never materialized.
// Matmuls: v_wmma_f32_16x16x32_bf16.  K/V/X tile staging: TDM
// (tensor_load_to_lds), double-buffered LDS shared by 4 waves per workgroup,
// synced with s_wait_tensorcnt + workgroup barriers.
// ---------------------------------------------------------------------------

typedef __attribute__((ext_vector_type(16))) __bf16 v16bf;
typedef __attribute__((ext_vector_type(8)))  __bf16 v8bf;
typedef __attribute__((ext_vector_type(8)))  float  v8f;
typedef __attribute__((ext_vector_type(4)))  unsigned int v4u;
typedef __attribute__((ext_vector_type(8)))  int v8i;
typedef __attribute__((ext_vector_type(4)))  int v4i;

#define BB 4
#define CC 128
#define NN 8192
#define MM 2048
#define NWAVES 4                     // waves per attention workgroup

// A-operand (16x32 bf16, MxK) per-lane K packing (ISA 7.12.2):
//   lanes 0-15 (hi=0): K = {0..7} U {16..23};  lanes 16-31 (hi=1): K = {8..15} U {24..31}
__device__ __forceinline__ int kmapA_base0(int hi) { return hi * 8; }
__device__ __forceinline__ int kmapA_base1(int hi) { return 16 + hi * 8; }
// B-operand (32x16 bf16, KxN) per-lane K packing: same as A (lanes 0-15: K={0..7}U{16..23}; 16-31: K={8..15}U{24..31})
__device__ __forceinline__ int kmapB_base(int hi) { return hi * 16; }

__device__ __forceinline__ v16bf load_fragA_bf16(const __bf16* row, int hi) {
  union { v16bf v; v8bf h[2]; } u;
  u.h[0] = *(const v8bf*)(row + kmapA_base0(hi));
  u.h[1] = *(const v8bf*)(row + kmapA_base1(hi));
  return u.v;
}
__device__ __forceinline__ v16bf load_fragB_bf16(const __bf16* row, int hi) {
  union { v16bf v; v8bf h[2]; } u;
  u.h[0] = *(const v8bf*)(row + kmapA_base0(hi));
  u.h[1] = *(const v8bf*)(row + kmapA_base1(hi));
  return u.v;
}

// Low 32 bits of the flat shared address = LDS byte offset (aperture mapping).
__device__ __forceinline__ unsigned lds_off_of(const void* p) {
  return (unsigned)(unsigned long long)(uintptr_t)p;
}

// ---------------------------------------------------------------------------
// TDM 2D tile load: Global -> LDS (packed row-major in LDS).
// D# per CDNA5 ISA:
//   group0: count=1 | lds_addr | global_addr[56:0] | type=2
//   group1: data_size | tensor_dim0/1 | tile_dim0/1 (tile_dim2=0) | dim0 stride
//   groups 2/3 + trailing group: zero (2D tensor), cpol = 0.
// data_size_code: 1 = 2 bytes (bf16), 2 = 4 bytes (f32). Dims in elements.
// Toolchain exposes the 6-arg builtin:
//   (uint32x4, int32x8, int32x4, int32x4, int32x8, i32 cpol)
// ---------------------------------------------------------------------------
__device__ __forceinline__ void tdm_load_2d(unsigned lds_off, const void* gptr,
                                            unsigned data_size_code,
                                            unsigned tile_d0, unsigned tile_d1,
                                            unsigned long long d0_stride) {
  unsigned long long ga = (unsigned long long)(uintptr_t)gptr;
  v4u g0;
  g0[0] = 1u;                                                  // count=1, user D#
  g0[1] = lds_off;                                             // lds_addr
  g0[2] = (unsigned)ga;                                        // global_addr[31:0]
  g0[3] = (unsigned)((ga >> 32) & 0x1ffffffu) | 0x80000000u;   // addr[56:32] | type=2
  v8i g1;
  g1[0] = (int)(data_size_code << 16);                         // data_size; no mcast/pad
  g1[1] = (int)((tile_d0 & 0xffffu) << 16);                    // tensor_dim0 = tile_d0
  g1[2] = (int)(((tile_d0 >> 16) & 0xffffu) |
                ((tile_d1 & 0xffffu) << 16));                  // tensor_dim1 = tile_d1
  g1[3] = (int)(((tile_d1 >> 16) & 0xffffu) |
                ((tile_d0 & 0xffffu) << 16));                  // tile_dim0
  g1[4] = (int)(tile_d1 & 0xffffu);                            // tile_dim1; tile_dim2=0
  g1[5] = (int)(d0_stride & 0xffffffffu);                      // dim0 stride [31:0]
  g1[6] = (int)((d0_stride >> 32) & 0xffffu);                  // dim0 stride [47:32]
  g1[7] = 0;
  v4i z4 = {0, 0, 0, 0};
  v8i z8 = {0, 0, 0, 0, 0, 0, 0, 0};
  __builtin_amdgcn_tensor_load_to_lds(g0, g1, z4, z4, z8, 0);
}

// ---------------------------------------------------------------------------
// Projection: O[o,p] = sum_c W[o,c] * X[b,c,p]   (1x1 conv)
//   mode 0: bf16 transposed outT[b][p][o]  (qT / kT)
//   mode 1: bf16 out[b][o][p]              (v)
//   mode 2: f32  out[b][o][p]              (skip -> d_out)
// One wave = one 16(o) x 16(p) tile. The strided X tile [128c x 16p] f32 is
// staged by one TDM descriptor (row 64B, stride P*4).
// ---------------------------------------------------------------------------
__global__ void proj_kernel(const float* __restrict__ W,
                            const float* __restrict__ X,
                            __bf16* __restrict__ out_bf,
                            float* __restrict__ out_f,
                            int P, int mode) {
  __shared__ float xsh[CC][16];     // 8 KB staged X tile
  const int lane = threadIdx.x & 31;
  const int lr   = lane & 15;
  const int hi   = lane >> 4;
  const int p0   = blockIdx.x * 16;
  const int o0   = blockIdx.y * 16;
  const int b    = blockIdx.z;
  const float* Xb = X + (size_t)b * CC * P;

  // TDM: X[c][p0..p0+16) for c = 0..127  ->  xsh[128][16]
  tdm_load_2d(lds_off_of(&xsh[0][0]), Xb + p0, /*f32*/2, 16, CC,
              (unsigned long long)P);
  __builtin_amdgcn_s_wait_tensorcnt(0);

  v8f acc = {};
#pragma unroll
  for (int kc = 0; kc < CC; kc += 32) {
    v16bf a, bf;
    const float* wrow = W + (size_t)(o0 + lr) * CC + kc;
#pragma unroll
    for (int i = 0; i < 8; ++i) a[i]     = (__bf16)wrow[kmapA_base0(hi) + i];
#pragma unroll
    for (int i = 0; i < 8; ++i) a[8 + i] = (__bf16)wrow[kmapA_base1(hi) + i];
#pragma unroll
    for (int i = 0; i < 8; ++i) bf[i]     = (__bf16)xsh[kc + kmapA_base0(hi) + i][lr];
#pragma unroll
    for (int i = 0; i < 8; ++i) bf[8 + i] = (__bf16)xsh[kc + kmapA_base1(hi) + i][lr];
    acc = __builtin_amdgcn_wmma_f32_16x16x32_bf16(false, a, false, bf,
                                                  (short)0, acc, false, false);
  }
#pragma unroll
  for (int r = 0; r < 8; ++r) {
    const int o = o0 + r + 8 * hi;
    const int p = p0 + lr;
    const float vv = acc[r];
    if (mode == 0)      out_bf[((size_t)b * P + p) * CC + o] = (__bf16)vv;
    else if (mode == 1) out_bf[((size_t)b * CC + o) * P + p] = (__bf16)vv;
    else                out_f [((size_t)b * CC + o) * P + p] = vv;
  }
}

// ---------------------------------------------------------------------------
// Fused attention, NWAVES waves per workgroup sharing TDM-staged K/V chunks.
// Each wave owns a 16-query-row tile; keys in chunks of 32 m-columns.
// Wave 0 issues the TDM double-buffer and waits TENSORcnt (per-wave counter);
// workgroup barriers publish / retire each LDS buffer.
//   qT [B][N][C] bf16, kT [B][M][C] bf16, vM [B][C][M] bf16
//   out [B][C][N] f32 pre-filled with skip; out += (softmax(qk/s) @ v^T)
// ---------------------------------------------------------------------------
__global__ void attn_kernel(const __bf16* __restrict__ qT,
                            const __bf16* __restrict__ kT,
                            const __bf16* __restrict__ vM,
                            float* __restrict__ out) {
  __shared__ __bf16 ktile[2][32][CC];       // 2 x 8 KB  (rows m, cols c)
  __shared__ __bf16 vtile[2][CC][32];       // 2 x 8 KB  (rows c, cols m)
  __shared__ __bf16 psh[NWAVES][16][32];    // per-wave P staging (D->A layout)
  const int lane = threadIdx.x & 31;
  const int wave = threadIdx.x >> 5;
  const int lr   = lane & 15;
  const int hi   = lane >> 4;
  const int n0   = (blockIdx.x * NWAVES + wave) * 16;
  const int b    = blockIdx.y;
  const float scale = 0.08838834764831845f;   // 1/sqrt(C)
  const int NCHUNK = MM / 32;

  // Q A-fragments (rows n0+lr, K=c), resident for the whole kernel
  v16bf qf[4];
  {
    const __bf16* qrow = qT + ((size_t)b * NN + n0 + lr) * CC;
#pragma unroll
    for (int kc = 0; kc < 4; ++kc) qf[kc] = load_fragA_bf16(qrow + kc * 32, hi);
  }

  float mrow[8], lrow[8];
  v8f oacc[8];
#pragma unroll
  for (int r = 0; r < 8; ++r) { mrow[r] = -__builtin_inff(); lrow[r] = 0.f; }
#pragma unroll
  for (int t = 0; t < 8; ++t) oacc[t] = {};

  // issue TDM loads for chunk ci into buffer buf (2 tensor ops, wave 0 only)
  auto issue = [&](int buf, int ci) {
    const int m0 = ci * 32;
    tdm_load_2d(lds_off_of(&ktile[buf][0][0]),
                kT + ((size_t)b * MM + m0) * CC, /*bf16*/1, CC, 32,
                (unsigned long long)CC);
    tdm_load_2d(lds_off_of(&vtile[buf][0][0]),
                vM + (size_t)b * CC * MM + m0, /*bf16*/1, 32, CC,
                (unsigned long long)MM);
  };

  if (wave == 0) issue(0, 0);                      // prefetch chunk 0
  for (int ci = 0; ci < NCHUNK; ++ci) {
    const int cur = ci & 1;
    if (wave == 0) {
      if (ci + 1 < NCHUNK) {
        issue(1 - cur, ci + 1);                    // prefetch next chunk
        __builtin_amdgcn_s_wait_tensorcnt(2);      // chunk ci's 2 loads done
      } else {
        __builtin_amdgcn_s_wait_tensorcnt(0);
      }
    }
    __syncthreads();                               // publish buffer `cur`

    // ---- energy tiles S0 (m cols 0..15 of chunk), S1 (16..31), from LDS
    v8f s0 = {}, s1 = {};
    const __bf16* krow0 = &ktile[cur][lr][0];
    const __bf16* krow1 = &ktile[cur][16 + lr][0];
#pragma unroll
    for (int kc = 0; kc < 4; ++kc) {
      v16bf kf0 = load_fragB_bf16(krow0 + kc * 32, hi);
      v16bf kf1 = load_fragB_bf16(krow1 + kc * 32, hi);
      s0 = __builtin_amdgcn_wmma_f32_16x16x32_bf16(false, qf[kc], false, kf0,
                                                   (short)0, s0, false, false);
      s1 = __builtin_amdgcn_wmma_f32_16x16x32_bf16(false, qf[kc], false, kf1,
                                                   (short)0, s1, false, false);
    }
    // ---- online softmax; row r+8*hi lives across the 16 lanes of this half
#pragma unroll
    for (int r = 0; r < 8; ++r) {
      float x0 = s0[r] * scale;
      float x1 = s1[r] * scale;
      float rm = fmaxf(x0, x1);
#pragma unroll
      for (int off = 1; off < 16; off <<= 1)
        rm = fmaxf(rm, __shfl_xor(rm, off, 32));
      const float mnew = fmaxf(mrow[r], rm);
      const float corr = __expf(mrow[r] - mnew);
      const float p0v  = __expf(x0 - mnew);
      const float p1v  = __expf(x1 - mnew);
      s0[r] = p0v; s1[r] = p1v;
      float rs = p0v + p1v;
#pragma unroll
      for (int off = 1; off < 16; off <<= 1)
        rs += __shfl_xor(rs, off, 32);
      lrow[r] = lrow[r] * corr + rs;
      mrow[r] = mnew;
#pragma unroll
      for (int t = 0; t < 8; ++t) oacc[t][r] *= corr;
    }
    // ---- D-layout -> A-layout via per-wave LDS tile (16n x 32m, bf16)
#pragma unroll
    for (int r = 0; r < 8; ++r) {
      psh[wave][r + 8 * hi][lr]      = (__bf16)s0[r];
      psh[wave][r + 8 * hi][16 + lr] = (__bf16)s1[r];
    }
    asm volatile("s_wait_dscnt 0" ::: "memory");   // DS in-order per wave; force it
    v16bf pf = load_fragA_bf16(&psh[wave][lr][0], hi);
    asm volatile("" ::: "memory");
    // ---- PV: out(n, c) += P(n, m) * v(c, m);  vtile rows = c, contiguous m
#pragma unroll
    for (int t = 0; t < 8; ++t) {
      v16bf vf = load_fragB_bf16(&vtile[cur][t * 16 + lr][0], hi);
      oacc[t] = __builtin_amdgcn_wmma_f32_16x16x32_bf16(false, pf, false, vf,
                                                        (short)0, oacc[t], false, false);
    }
    __syncthreads();   // all waves done with buffer `cur` before its next TDM
  }

  // ---- epilogue: normalize and add skip (already resident in out)
  float inv[8];
#pragma unroll
  for (int r = 0; r < 8; ++r) inv[r] = 1.0f / lrow[r];
#pragma unroll
  for (int t = 0; t < 8; ++t) {
    const int c = t * 16 + lr;
    float* po = out + ((size_t)b * CC + c) * NN + n0 + 8 * hi;
#pragma unroll
    for (int r = 0; r < 8; ++r) po[r] += oacc[t][r] * inv[r];
  }
}

// ---------------------------------------------------------------------------
extern "C" void kernel_launch(void* const* d_in, const int* in_sizes, int n_in,
                              void* d_out, int out_size, void* d_ws, size_t ws_size,
                              hipStream_t stream) {
  (void)in_sizes; (void)n_in; (void)out_size; (void)ws_size;
  const float* pcd_up   = (const float*)d_in[0];
  const float* pcd_down = (const float*)d_in[1];
  const float* wq       = (const float*)d_in[2];
  const float* wk       = (const float*)d_in[3];
  const float* wv       = (const float*)d_in[4];
  const float* wskip    = (const float*)d_in[5];
  float* out = (float*)d_out;

  // workspace: qT (8 MB) + kT (2 MB) + v (2 MB), all bf16
  __bf16* qT = (__bf16*)d_ws;
  __bf16* kT = qT + (size_t)BB * NN * CC;
  __bf16* vM = kT + (size_t)BB * MM * CC;

  dim3 blk(32);
  proj_kernel<<<dim3(NN / 16, CC / 16, BB), blk, 0, stream>>>(wq,    pcd_up,   qT, nullptr, NN, 0);
  proj_kernel<<<dim3(MM / 16, CC / 16, BB), blk, 0, stream>>>(wk,    pcd_down, kT, nullptr, MM, 0);
  proj_kernel<<<dim3(MM / 16, CC / 16, BB), blk, 0, stream>>>(wv,    pcd_down, vM, nullptr, MM, 1);
  proj_kernel<<<dim3(NN / 16, CC / 16, BB), blk, 0, stream>>>(wskip, pcd_up,   nullptr, out, NN, 2);
  attn_kernel<<<dim3(NN / (16 * NWAVES), BB), dim3(32 * NWAVES), 0, stream>>>(qT, kT, vM, out);
}
